// LinearAttention_59107339927644
// MI455X (gfx1250) — matrix operand order repaired and checked by reading, not compiled, not run on hardware
//
#include <hip/hip_runtime.h>

// ---------------------------------------------------------------------------
// LinearAttention for MI455X (gfx1250, wave32, WMMA bf16).
//
// Shapes: b=1, C=512, T=16, 64x64 spatial -> N=4096, S=65536, heads=8, d=64.
//
// Pipeline:
//   0) convert x, w_qkv, w_out to bf16 in workspace (vectorized)
//   1) QKV GEMM (bf16 WMMA, f32 acc, 128x64 tiles) -> qkv_bf[1536][65536]
//   2) softmax over d on Q rows (*0.125), softmax over n on K rows (in place)
//   3) per (t,head): context = K*V^T (WMMA), out = context^T * Q (WMMA)
//   4) out-proj GEMM (bf16 WMMA) + bias -> f32 d_out
//
// Register blocking: each wave owns 2 M-strips x 4 N-tiles (8 f32 accum
// tiles = 64 VGPRs) -> 1.5 b128 LDS reads / WMMA. All global staging is b128.
// ---------------------------------------------------------------------------

typedef __attribute__((ext_vector_type(16))) __bf16 v16bf;
typedef __attribute__((ext_vector_type(8)))  float  v8f;

union Frag {
    v16bf v;
    uint4 q[2];
};
union U4BF8 {           // 8 bf16 <-> one b128
    uint4  q;
    __bf16 h[8];
};
union PK4 {             // 4 bf16 <-> one b64
    uint2  u;
    __bf16 h[4];
};

__device__ __forceinline__ v8f bfwmma(v16bf a, v16bf b, v8f c) {
    // D = A(16x32 bf16) * B(32x16 bf16) + C(16x16 f32)
    return __builtin_amdgcn_wmma_f32_16x16x32_bf16(
        /*neg_a=*/false, a, /*neg_b=*/false, b,
        /*c_mod=*/(short)0, c, /*reuse_a=*/false, /*reuse_b=*/false);
}

#define SCOLS 65536     // t*4096 + xy
#define NSPAT 4096
#define CDIM  512
#define O3    1536
#define HEADS 8
#define DHEAD 64
#define TDIM  16

// ---------------------------------------------------------------------------
// f32 -> bf16 conversion, 4 elems/thread (b128 in, b64 out)
// ---------------------------------------------------------------------------
__global__ void cvt_f32_bf16(const float4* __restrict__ src,
                             uint2* __restrict__ dst, int n4) {
    int i = blockIdx.x * blockDim.x + threadIdx.x;
    int stride = gridDim.x * blockDim.x;
    for (; i < n4; i += stride) {
        float4 f = src[i];
        PK4 p;
        p.h[0] = (__bf16)f.x; p.h[1] = (__bf16)f.y;
        p.h[2] = (__bf16)f.z; p.h[3] = (__bf16)f.w;
        dst[i] = p.u;
    }
}

// ---------------------------------------------------------------------------
// 128x64-output-tile GEMM core (4 waves, each wave: 2 M-strips x 4 N-tiles):
//   Y[ob+0..127][sb+0..63] = sum_c W[o][c] * X[c][s]
// W: [rowsW][512] bf16 row-major, X: [512][SCOLS] bf16 row-major.
// ---------------------------------------------------------------------------
#define TPAD 40  // padded LDS row stride (elems); keeps b128 runs 16B aligned

template <bool BIAS, typename OutT>
__device__ __forceinline__
void gemm_core(const __bf16* __restrict__ W, const __bf16* __restrict__ X,
               const float* __restrict__ bias, OutT* __restrict__ Y) {
    __shared__ __bf16 Wt[128 * TPAD];  // [o_local][c_local]
    __shared__ __bf16 Xt[64 * TPAD];   // [s_local][c_local] (transposed stage)
    const int tid  = threadIdx.x;
    const int lane = tid & 31, wv = tid >> 5;
    const int ln   = lane & 15, hl = lane >> 4;
    const int sb   = blockIdx.x * 64;
    const int ob   = blockIdx.y * 128;

    v8f acc[2][4] = {};

    for (int kc = 0; kc < CDIM; kc += 32) {
        __syncthreads();
        // W tile [128][32]: 512 b128 transfers, b128 on both sides
#pragma unroll
        for (int u = tid; u < 512; u += 128) {
            int r = u >> 2, c8 = (u & 3) * 8;
            *(uint4*)&Wt[r * TPAD + c8] =
                *(const uint4*)&W[(size_t)(ob + r) * CDIM + kc + c8];
        }
        // X tile [32][64] -> transposed: b128 global, scalar LDS scatter
#pragma unroll
        for (int u = tid; u < 256; u += 128) {
            int c = u >> 3, s8 = (u & 7) * 8;
            U4BF8 v;
            v.q = *(const uint4*)&X[(size_t)(kc + c) * SCOLS + sb + s8];
#pragma unroll
            for (int i = 0; i < 8; ++i) Xt[(s8 + i) * TPAD + c] = v.h[i];
        }
        __syncthreads();
        if (kc + 32 < CDIM)
            __builtin_prefetch(
                &X[(size_t)(kc + 32 + (tid >> 6)) * SCOLS + sb + (tid & 63)], 0, 1);

        Frag a[2];
#pragma unroll
        for (int mi = 0; mi < 2; ++mi) {
            const __bf16* ap = &Wt[(wv * 32 + mi * 16 + ln) * TPAD + hl * 8];
            a[mi].q[0] = *(const uint4*)(ap);
            a[mi].q[1] = *(const uint4*)(ap + 16);
        }
#pragma unroll
        for (int nt = 0; nt < 4; ++nt) {
            Frag b;
            const __bf16* bp = &Xt[(nt * 16 + ln) * TPAD + hl * 8];
            b.q[0] = *(const uint4*)(bp);
            b.q[1] = *(const uint4*)(bp + 16);
            acc[0][nt] = bfwmma(a[0].v, b.v, acc[0][nt]);
            acc[1][nt] = bfwmma(a[1].v, b.v, acc[1][nt]);
        }
    }

#pragma unroll
    for (int mi = 0; mi < 2; ++mi)
#pragma unroll
        for (int nt = 0; nt < 4; ++nt)
#pragma unroll
            for (int r = 0; r < 8; ++r) {
                int row = ob + wv * 32 + mi * 16 + hl * 8 + r;
                int col = sb + nt * 16 + ln;
                float v = acc[mi][nt][r];
                if (BIAS) v += bias[row];
                Y[(size_t)row * SCOLS + col] = (OutT)v;
            }
}

__global__ __launch_bounds__(128)
void gemm_qkv(const __bf16* __restrict__ W, const __bf16* __restrict__ X,
              __bf16* __restrict__ Y) {
    gemm_core<false, __bf16>(W, X, nullptr, Y);
}

__global__ __launch_bounds__(128)
void gemm_out(const __bf16* __restrict__ W, const __bf16* __restrict__ X,
              const float* __restrict__ bias, float* __restrict__ Y) {
    gemm_core<true, float>(W, X, bias, Y);
}

// ---------------------------------------------------------------------------
// softmax over d (=64 rows of a head) for Q, per (head, s) column; * SCALE.
// Q rows live at qkv[(head*64 + d)*SCOLS + s]; coalesced across threads.
// ---------------------------------------------------------------------------
__global__ void softmax_q(__bf16* __restrict__ qkv) {
    int gid = blockIdx.x * blockDim.x + threadIdx.x;       // HEADS*SCOLS threads
    if (gid >= HEADS * SCOLS) return;
    int head = gid >> 16;
    int s    = gid & (SCOLS - 1);
    size_t base = (size_t)(head * DHEAD) * SCOLS + s;
    float v[DHEAD];
    float m = -1e30f;
#pragma unroll
    for (int d = 0; d < DHEAD; ++d) {
        v[d] = (float)qkv[base + (size_t)d * SCOLS];
        m = fmaxf(m, v[d]);
    }
    float sum = 0.f;
#pragma unroll
    for (int d = 0; d < DHEAD; ++d) {
        v[d] = __expf(v[d] - m);
        sum += v[d];
    }
    float inv = 0.125f / sum;                               // SCALE = 64^-0.5
#pragma unroll
    for (int d = 0; d < DHEAD; ++d)
        qkv[base + (size_t)d * SCOLS] = (__bf16)(v[d] * inv);
}

// ---------------------------------------------------------------------------
// softmax over n (=4096 within one t) for K rows; one wave per (t,head,d).
// K rows live at qkv[(512 + head*64 + d)*SCOLS + t*4096 + n].
// b128 loads: lane covers 8 consecutive elems, wave covers 256 per pass.
// ---------------------------------------------------------------------------
__global__ __launch_bounds__(128)
void softmax_k(__bf16* __restrict__ qkv) {
    int wid  = (blockIdx.x * blockDim.x + threadIdx.x) >> 5;  // 8192 waves
    int lane = threadIdx.x & 31;
    int t    = wid >> 9;
    int rem  = wid & 511;
    int head = rem >> 6;
    int d    = rem & 63;
    size_t base = (size_t)(CDIM + head * DHEAD + d) * SCOLS + (size_t)t * NSPAT;

    float m = -1e30f;
    for (int i = lane * 8; i < NSPAT; i += 256) {
        U4BF8 v; v.q = *(const uint4*)&qkv[base + i];
#pragma unroll
        for (int j = 0; j < 8; ++j) m = fmaxf(m, (float)v.h[j]);
    }
#pragma unroll
    for (int off = 16; off > 0; off >>= 1)
        m = fmaxf(m, __shfl_xor(m, off, 32));

    float sum = 0.f;
    for (int i = lane * 8; i < NSPAT; i += 256) {
        U4BF8 v; v.q = *(const uint4*)&qkv[base + i];
#pragma unroll
        for (int j = 0; j < 8; ++j) sum += __expf((float)v.h[j] - m);
    }
#pragma unroll
    for (int off = 16; off > 0; off >>= 1)
        sum += __shfl_xor(sum, off, 32);

    float inv = 1.f / sum;
    for (int i = lane * 8; i < NSPAT; i += 256) {
        U4BF8 v; v.q = *(const uint4*)&qkv[base + i];
        U4BF8 o;
#pragma unroll
        for (int j = 0; j < 8; ++j)
            o.h[j] = (__bf16)(__expf((float)v.h[j] - m) * inv);
        *(uint4*)&qkv[base + i] = o.q;
    }
}

// ---------------------------------------------------------------------------
// Linear attention per (t, head): one 128-thread block each (128 blocks).
//  Phase A: ctx[d][e] = sum_n K'[d][n] * V[e][n]      (WMMA over n)
//  Phase B: out[e][n] = sum_d ctx[d][e] * Q'[d][n]    (WMMA over d)
// ---------------------------------------------------------------------------
#define QPAD 72   // bf16 row stride for transposed Q block (16B-aligned runs)

__global__ __launch_bounds__(128)
void attn(const __bf16* __restrict__ qkv, __bf16* __restrict__ aout) {
    __shared__ float  ctxT[64 * 65];     // [e][d] = context[d][e]
    __shared__ __bf16 Qt[64 * QPAD];     // [n_local][d]
    const int tid  = threadIdx.x;
    const int lane = tid & 31, wv = tid >> 5;
    const int ln   = lane & 15, hl = lane >> 4;
    const int t    = blockIdx.x >> 3;
    const int head = blockIdx.x & 7;

    const size_t colbase = (size_t)t * NSPAT;
    const size_t qbase = (size_t)(head * DHEAD) * SCOLS + colbase;
    const size_t kbase = (size_t)(CDIM + head * DHEAD) * SCOLS + colbase;
    const size_t vbase = (size_t)(2 * CDIM + head * DHEAD) * SCOLS + colbase;
    const size_t obase = qbase;  // aout rows: head*64 + e

    // ---- Phase A: context accumulation over n -----------------------------
    v8f c[4] = {};
    const int m = wv * 16 + ln;                       // d-row handled by lane
    for (int kc = 0; kc < NSPAT; kc += 32) {
        Frag a;
        const __bf16* ap = &qkv[kbase + (size_t)m * SCOLS + kc + hl * 8];
        a.q[0] = *(const uint4*)(ap);
        a.q[1] = *(const uint4*)(ap + 16);
#pragma unroll
        for (int et = 0; et < 4; ++et) {
            Frag b;
            const __bf16* bp =
                &qkv[vbase + (size_t)(et * 16 + ln) * SCOLS + kc + hl * 8];
            b.q[0] = *(const uint4*)(bp);
            b.q[1] = *(const uint4*)(bp + 16);
            c[et] = bfwmma(a.v, b.v, c[et]);
        }
    }

    // scatter context transposed to LDS: ctxT[e][d]
#pragma unroll
    for (int et = 0; et < 4; ++et)
#pragma unroll
        for (int r = 0; r < 8; ++r)
            ctxT[(et * 16 + ln) * 65 + (wv * 16 + hl * 8 + r)] = c[et][r];
    __syncthreads();

    // A' = context^T fragments (per-wave e-strip), f32 -> bf16
    Frag af[2];
#pragma unroll
    for (int q = 0; q < 2; ++q)
#pragma unroll
        for (int j = 0; j < 16; ++j) {
            int k = q * 32 + (j >> 3) * 16 + hl * 8 + (j & 7);   // d index
            af[q].v[j] = (__bf16)ctxT[(wv * 16 + ln) * 65 + k];
        }

    // ---- Phase B: read-out over n -----------------------------------------
    for (int nb = 0; nb < NSPAT; nb += 64) {
        __syncthreads();
        // stage Q block [64 d][64 n] transposed: b128 global, scalar scatter
#pragma unroll
        for (int u = tid; u < 512; u += 128) {
            int d = u >> 3, n8 = (u & 7) * 8;
            U4BF8 v;
            v.q = *(const uint4*)&qkv[qbase + (size_t)d * SCOLS + nb + n8];
#pragma unroll
            for (int i = 0; i < 8; ++i) Qt[(n8 + i) * QPAD + d] = v.h[i];
        }
        __syncthreads();
#pragma unroll
        for (int sub = 0; sub < 4; ++sub) {
            v8f o = {};
#pragma unroll
            for (int q = 0; q < 2; ++q) {
                Frag b;
                const __bf16* bp = &Qt[(sub * 16 + ln) * QPAD + q * 32 + hl * 8];
                b.q[0] = *(const uint4*)(bp);
                b.q[1] = *(const uint4*)(bp + 16);
                o = bfwmma(af[q].v, b.v, o);
            }
#pragma unroll
            for (int r = 0; r < 8; ++r) {
                int e   = wv * 16 + hl * 8 + r;
                int col = nb + sub * 16 + ln;
                aout[obase + (size_t)e * SCOLS + col] = (__bf16)o[r];
            }
        }
    }
}

// ---------------------------------------------------------------------------
// Launcher
// ---------------------------------------------------------------------------
extern "C" void kernel_launch(void* const* d_in, const int* in_sizes, int n_in,
                              void* d_out, int out_size, void* d_ws, size_t ws_size,
                              hipStream_t stream) {
    const float* x     = (const float*)d_in[0];   // [512][65536]
    const float* w_qkv = (const float*)d_in[1];   // [1536][512]
    const float* w_out = (const float*)d_in[2];   // [512][512]
    const float* b_out = (const float*)d_in[3];   // [512]
    float* out = (float*)d_out;                   // [512][65536]

    // Workspace layout (bf16):
    //   qkv_bf  [1536][65536]  201.3 MB   (Q rows 0..511, K 512..1023, V 1024..1535)
    //   aout_bf [ 512][65536]   67.1 MB
    //   x_bf    [ 512][65536]   67.1 MB
    //   wqkv_bf [1536][512]      1.5 MB
    //   wout_bf [ 512][512]      0.5 MB
    __bf16* qkv_bf  = (__bf16*)d_ws;
    __bf16* aout_bf = qkv_bf  + (size_t)O3 * SCOLS;
    __bf16* x_bf    = aout_bf + (size_t)CDIM * SCOLS;
    __bf16* wqkv_bf = x_bf    + (size_t)CDIM * SCOLS;
    __bf16* wout_bf = wqkv_bf + (size_t)O3 * CDIM;

    // 0) conversions (4 elems/thread, vectorized)
    cvt_f32_bf16<<<4096, 256, 0, stream>>>((const float4*)x, (uint2*)x_bf,
                                           (CDIM * SCOLS) / 4);
    cvt_f32_bf16<<<768, 256, 0, stream>>>((const float4*)w_qkv, (uint2*)wqkv_bf,
                                          (O3 * CDIM) / 4);
    cvt_f32_bf16<<<256, 256, 0, stream>>>((const float4*)w_out, (uint2*)wout_bf,
                                          (CDIM * CDIM) / 4);

    // 1) QKV projection: 1024 s-tiles x 12 o-tiles (128-row tiles)
    gemm_qkv<<<dim3(SCOLS / 64, O3 / 128), 128, 0, stream>>>(wqkv_bf, x_bf, qkv_bf);

    // 2) softmaxes (in place on qkv_bf)
    softmax_q<<<(HEADS * SCOLS) / 256, 256, 0, stream>>>(qkv_bf);
    softmax_k<<<2048, 128, 0, stream>>>(qkv_bf);   // 8192 waves, 1 per (t,head,d)

    // 3) attention per (t, head)
    attn<<<TDIM * HEADS, 128, 0, stream>>>(qkv_bf, aout_bf);

    // 4) output projection + bias
    gemm_out<<<dim3(SCOLS / 64, CDIM / 128), 128, 0, stream>>>(wout_bf, aout_bf, b_out, out);
}
